// Kernel_63058709840173
// MI455X (gfx1250) — matrix operand order repaired and checked by reading, not compiled, not run on hardware
//
#include <hip/hip_runtime.h>
#include <hip/hip_bf16.h>
#include <stdint.h>

// Batched MLP (7->90->90->90->1, relu) over 864,000 rows, then sum over DEV=5.
// Compute-bound (~29 GFLOP on ~1MB of global I/O) -> route the 90x90 layers
// through v_wmma_f32_16x16x32_f16 (f16 in, f32 accum). K padded 90->96 (3 k-chunks
// of 32), N padded 90->96 (6 n-tiles of 16). Weights pre-swizzled into WMMA
// B-fragment order so fragment loads are contiguous ds_load_b128s.

typedef __attribute__((ext_vector_type(16))) _Float16 v16h;
typedef __attribute__((ext_vector_type(8)))  float    v8f;

#define N_B      8
#define N_OC     6
#define N_DEV    5
#define N_HOUT   60
#define N_L      (N_HOUT * N_HOUT)          // 3600
#define N_ROWS   (N_B * N_OC * N_L * N_DEV) // 864000
#define NPAD     96
#define FRAG_DW  4608                       // 6 ntiles * 3 kchunks * 32 lanes * 8 dwords
#define TILES    (N_ROWS / 16)              // 54000
#define WAVES_PB 8
#define BLOCKS   (TILES / WAVES_PB)         // 6750

// ---- prep: f32 [90][90] row-major -> padded f16 B-fragments -------------------
// frag[((nt*3+kc)*32 + lane)*8 + v] packs B[k0][col] (lo) and B[k0+1][col] (hi)
// with k0 = kc*32 + (lane/16)*16 + 2v, col = nt*16 + lane%16 (zeros past 90).
__global__ void prep_frags_kernel(const float* __restrict__ W1,
                                  const float* __restrict__ W2,
                                  uint32_t* __restrict__ frag) {
  int tid = blockIdx.x * 256 + threadIdx.x;
  if (tid >= 2 * FRAG_DW) return;
  const float* W = (tid < FRAG_DW) ? W1 : W2;
  uint32_t* dst = frag + ((tid < FRAG_DW) ? 0 : FRAG_DW);
  int rem  = (tid < FRAG_DW) ? tid : (tid - FRAG_DW);
  int v    = rem & 7;
  int lane = (rem >> 3) & 31;
  int kc   = (rem >> 8) % 3;
  int nt   = rem / 768;
  int np    = lane & 15;
  int khalf = lane >> 4;
  int k0  = kc * 32 + khalf * 16 + 2 * v;
  int col = nt * 16 + np;
  float lo = (k0     < 90 && col < 90) ? W[k0 * 90 + col]       : 0.0f;
  float hi = (k0 + 1 < 90 && col < 90) ? W[(k0 + 1) * 90 + col] : 0.0f;
  _Float16 hlo = (_Float16)lo;
  _Float16 hhi = (_Float16)hi;
  uint32_t plo, phi;
  __builtin_memcpy(&plo, &hlo, 2);
  __builtin_memcpy(&phi, &hhi, 2);
  dst[rem] = ((phi & 0xFFFFu) << 16) | (plo & 0xFFFFu);
}

__global__ void zero_kernel(float* __restrict__ p, int n) {
  int i = blockIdx.x * 256 + threadIdx.x;
  if (i < n) p[i] = 0.0f;
}

// ---- fused MLP kernel ---------------------------------------------------------
__launch_bounds__(256)
__global__ void mlp_kernel(const float* __restrict__ x,     // [8,1,64,64]
                           const float* __restrict__ ctrl,  // [6,1,5,2]
                           const float* __restrict__ W0,    // [7,90]
                           const float* __restrict__ b0,    // [90]
                           const float* __restrict__ b1,    // [90]
                           const float* __restrict__ b2,    // [90]
                           const float* __restrict__ W3,    // [90,1]
                           const float* __restrict__ b3,    // [1]
                           const uint32_t* __restrict__ frags, // 2*FRAG_DW dwords
                           float* __restrict__ out) {       // [8,6,60,60] (zeroed)
  __shared__ uint32_t ldsW[2 * FRAG_DW];          // 36 KB: W1,W2 fragments
  __shared__ _Float16 hA[WAVES_PB][16 * NPAD];    // 24 KB: per-wave activation slab

  const int t = threadIdx.x;
  for (int i = t; i < 2 * FRAG_DW; i += 256) ldsW[i] = frags[i];
  __syncthreads();

  const int wave = t >> 5;
  const int lane = t & 31;
  const int np    = lane & 15;
  const int khalf = lane >> 4;
  const int tile    = blockIdx.x * WAVES_PB + wave;
  const int rowBase = tile * 16;
  _Float16* h = &hA[wave][0];

  // ---- layer 0: build 7 electrode values, relu(in @ W0 + b0) -> h (f16) ----
  if (lane < 16) {
    int r  = rowBase + lane;
    int dev = r % N_DEV;   int q1 = r / N_DEV;
    int l   = q1 % N_L;    int q2 = q1 / N_L;
    int oc  = q2 % N_OC;   int b  = q2 / N_OC;
    int i0 = l / N_HOUT, j0 = l % N_HOUT;
    const float* xr = x + ((size_t)b * 64 + (i0 + dev)) * 64 + j0;
    // electrode order: [xw1,xw2,xw3,xw4, ctrl0, xw0, ctrl1]
    float e0 = xr[1], e1 = xr[2], e2 = xr[3], e3 = xr[4], e5 = xr[0];
    const float* cp = ctrl + ((size_t)oc * N_DEV + dev) * 2;
    float e4 = cp[0], e6 = cp[1];
    for (int n = 0; n < NPAD; ++n) {
      float a = 0.0f;
      if (n < 90) {
        a = b0[n]
          + e0 * W0[0 * 90 + n] + e1 * W0[1 * 90 + n] + e2 * W0[2 * 90 + n]
          + e3 * W0[3 * 90 + n] + e4 * W0[4 * 90 + n] + e5 * W0[5 * 90 + n]
          + e6 * W0[6 * 90 + n];
        a = fmaxf(a, 0.0f);
      }
      h[lane * NPAD + n] = (_Float16)a;
    }
  }
  __syncthreads();

  // ---- layers 1 & 2: 16x96 @ 96x96 via v_wmma_f32_16x16x32_f16 ----
  for (int layer = 0; layer < 2; ++layer) {
    const uint32_t* wf = ldsW + layer * FRAG_DW;
    const float* bias = (layer == 0) ? b1 : b2;

    v8f acc[6];
    #pragma unroll
    for (int nt = 0; nt < 6; ++nt) {
      int col = nt * 16 + np;
      float bv = (col < 90) ? bias[col] : 0.0f;
      #pragma unroll
      for (int v = 0; v < 8; ++v) acc[nt][v] = bv;
    }

    #pragma unroll
    for (int kc = 0; kc < 3; ++kc) {
      // A fragment: row np, K = kc*32 + khalf*16 + {0..15} in v[0..7]
      // (matches B fragment k order) -> two contiguous 16B LDS reads.
      union { v16h v; uint4 u[2]; } A;
      const uint4* ap = (const uint4*)(h + np * NPAD + kc * 32 + khalf * 16);
      A.u[0] = ap[0];   // + 0 halves
      A.u[1] = ap[1];   // + 8 halves (16 bytes)
      #pragma unroll
      for (int nt = 0; nt < 6; ++nt) {
        union { v16h v; uint4 u[2]; } Bm;
        const uint4* bp = (const uint4*)(wf + ((size_t)((nt * 3 + kc) * 32 + lane)) * 8);
        Bm.u[0] = bp[0];
        Bm.u[1] = bp[1];
        acc[nt] = __builtin_amdgcn_wmma_f32_16x16x32_f16(
            false, A.v, false, Bm.v, (short)0, acc[nt], false, false);
      }
    }
    __syncthreads();  // all A-fragment reads done before slab overwrite

    // C layout: VGPR v -> M = v + khalf*8, N = np  -> relu -> f16 back to slab
    #pragma unroll
    for (int nt = 0; nt < 6; ++nt) {
      int col = nt * 16 + np;
      #pragma unroll
      for (int v = 0; v < 8; ++v) {
        float val = fmaxf(acc[nt][v], 0.0f);
        h[(khalf * 8 + v) * NPAD + col] = (_Float16)val;
      }
    }
    __syncthreads();
  }

  // ---- layer 3: 90->1 dot + sum over DEV via atomicAdd ----
  if (lane < 16) {
    float acc = b3[0];
    for (int k = 0; k < 90; ++k)
      acc += (float)h[lane * NPAD + k] * W3[k];
    int r = rowBase + lane;
    atomicAdd(&out[r / N_DEV], acc);
  }
}

extern "C" void kernel_launch(void* const* d_in, const int* in_sizes, int n_in,
                              void* d_out, int out_size, void* d_ws, size_t ws_size,
                              hipStream_t stream) {
  const float* x    = (const float*)d_in[0];
  const float* ctrl = (const float*)d_in[1];
  const float* W0   = (const float*)d_in[2];
  const float* b0   = (const float*)d_in[3];
  const float* W1   = (const float*)d_in[4];
  const float* b1   = (const float*)d_in[5];
  const float* W2   = (const float*)d_in[6];
  const float* b2   = (const float*)d_in[7];
  const float* W3   = (const float*)d_in[8];
  const float* b3   = (const float*)d_in[9];
  float* out = (float*)d_out;
  uint32_t* frag = (uint32_t*)d_ws;   // 2*FRAG_DW dwords = 36 KB

  prep_frags_kernel<<<(2 * FRAG_DW + 255) / 256, 256, 0, stream>>>(W1, W2, frag);
  zero_kernel<<<(out_size + 255) / 256, 256, 0, stream>>>(out, out_size);
  mlp_kernel<<<BLOCKS, 256, 0, stream>>>(x, ctrl, W0, b0, b1, b2, W3, b3, frag, out);
}
